// CrossAttention_63797444215108
// MI455X (gfx1250) — matrix operand order repaired and checked by reading, not compiled, not run on hardware
//
#include <hip/hip_runtime.h>
#include <hip/hip_bf16.h>

#define DEVFN static __device__ __forceinline__

typedef __bf16 bf16_t;
typedef __attribute__((ext_vector_type(16))) __bf16 v16bf;
typedef __attribute__((ext_vector_type(8)))  float  v8f;
typedef __attribute__((ext_vector_type(4)))  unsigned int u32x4;
typedef unsigned short bfraw;

constexpr int Bb = 4, Ss = 2048, Cc = 1024, Hh = 16, HSz = 64, Mm = 3;
constexpr int DM = Hh * HSz;   // 1024
constexpr int CH = Cc / 2;     // 512

// ---- bf16 helpers: native hardware converts ----
DEVFN bfraw f2br(float f) {
  bf16_t b = (bf16_t)f;           // v_cvt f32->bf16 (RNE)
  bfraw r; __builtin_memcpy(&r, &b, 2);
  return r;
}
DEVFN unsigned pk2(float a, float b) {
  return (unsigned)f2br(a) | ((unsigned)f2br(b) << 16);
}

DEVFN v8f wmma_bf16(v16bf a, v16bf b, v8f c) {
  return __builtin_amdgcn_wmma_f32_16x16x32_bf16(false, a, false, b, (short)0, c,
                                                 false, false);
}

// 16 contiguous bf16 -> fragment (two b128 loads)
DEVFN v16bf frag_c(const bfraw* p) {
  union { u32x4 u[2]; v16bf v; } x;
  const u32x4* q = (const u32x4*)p;
  x.u[0] = q[0]; x.u[1] = q[1];
  return x.v;
}
// 8 contiguous bf16 at p, 8 at p+16 -> fragment (two b128 loads)
DEVFN v16bf frag_s(const bfraw* p) {
  union { u32x4 u[2]; v16bf v; } x;
  const u32x4* q = (const u32x4*)p;
  x.u[0] = q[0]; x.u[1] = q[2];
  return x.v;
}

// A fragment 16x32 from bf16 row-major [*, lda] (contiguous k order: k = 16h + s)
DEVFN v16bf load_a(const bfraw* A, int lda, int row0, int k0, int lane) {
  return frag_c(A + (size_t)(row0 + (lane & 15)) * lda + k0 +
                ((lane < 16) ? 0 : 16));
}
// B fragment from K-panel packed weights: Wp[(k/32)*N*32 + n*32 + (k&31)]
DEVFN v16bf load_bp(const bfraw* Wp, int N, int k0, int n0, int lane) {
  return frag_c(Wp + ((size_t)(k0 >> 5) * N + n0 + (lane & 15)) * 32 +
                ((lane < 16) ? 0 : 16));
}
// B fragment for K^T: logical B(d,key) = Kmat[key*HS + d] (contiguous in d)
DEVFN v16bf load_kT(const bfraw* Km, int key0, int d0, int lane) {
  return frag_c(Km + (size_t)(key0 + (lane & 15)) * HSz + d0 +
                ((lane < 16) ? 0 : 16));
}
// B fragment from V stored in key-panel layout: Vp[(s/32)*HS*32 + d*32 + (s&31)]
DEVFN v16bf load_vp(const bfraw* Vp, int kt, int d0, int lane) {
  return frag_c(Vp + (size_t)(kt >> 5) * (HSz * 32) + (d0 + (lane & 15)) * 32 +
                ((lane < 16) ? 0 : 16));
}

// ---------- pack f32 -> bf16 row-major (8 elems/thread, b128 in / b128 out) ----------
__global__ __launch_bounds__(256) void pack_x_kernel(const float* __restrict__ X,
                                                     bfraw* __restrict__ Y, long n) {
  long i = ((long)blockIdx.x * blockDim.x + threadIdx.x) * 8;
  if (i >= n) return;
  const float4* p4 = reinterpret_cast<const float4*>(X + i);
  float4 f0 = p4[0], f1 = p4[1];
  uint4 o;
  o.x = pk2(f0.x, f0.y); o.y = pk2(f0.z, f0.w);
  o.z = pk2(f1.x, f1.y); o.w = pk2(f1.z, f1.w);
  *reinterpret_cast<uint4*>(Y + i) = o;
}

// ---------- pack f32 weights -> bf16 K-panel layout [K/32][N][32] ----------
__global__ __launch_bounds__(256) void pack_w_kernel(const float* __restrict__ W0,
                                                     bfraw* __restrict__ Y0,
                                                     int ldw, int colOff, int K, int N,
                                                     long hsW, long hsY) {
  long idx = (long)blockIdx.x * blockDim.x + threadIdx.x;
  if (idx >= (long)K * N) return;
  int k = (int)(idx / N), n = (int)(idx % N);
  const float* W = W0 + (long)blockIdx.y * hsW;
  bfraw* Y = Y0 + (long)blockIdx.y * hsY;
  Y[((long)(k >> 5) * N + n) * 32 + (k & 31)] = f2br(W[(long)k * ldw + colOff + n]);
}

// ---------- projection: Y(b,h,s,d) = Xbf(b,s,:) @ Wp(h); row-major or key-panel out ----------
__global__ __launch_bounds__(128) void proj_kernel(const bfraw* __restrict__ X0,
                                                   const bfraw* __restrict__ Wp0,
                                                   bfraw* __restrict__ Y0,
                                                   int panelOut) {
  const int lane = threadIdx.x & 31;
  const int wid  = threadIdx.x >> 5;
  const int h = blockIdx.y, b = blockIdx.z;
  const int s0 = (blockIdx.x * 4 + wid) * 16;
  const bfraw* X  = X0 + (size_t)b * Ss * Cc;
  const bfraw* Wp = Wp0 + (size_t)h * Cc * HSz;
  bfraw* Y = Y0 + (size_t)(b * Hh + h) * Ss * HSz;

  v8f acc[4] = {};
  for (int k0 = 0; k0 < Cc; k0 += 32) {
    v16bf a = load_a(X, Cc, s0, k0, lane);
#pragma unroll
    for (int t = 0; t < 4; ++t)
      acc[t] = wmma_bf16(a, load_bp(Wp, HSz, k0, t * 16, lane), acc[t]);
  }
  const int n  = lane & 15;
  const int rb = (lane < 16) ? 0 : 8;
#pragma unroll
  for (int t = 0; t < 4; ++t)
#pragma unroll
    for (int r = 0; r < 8; ++r) {
      const int s = s0 + rb + r, d = t * 16 + n;
      const size_t idx = panelOut
          ? (size_t)(s >> 5) * (HSz * 32) + (size_t)d * 32 + (s & 31)
          : (size_t)s * HSz + d;
      Y[idx] = f2br(acc[t][r]);
    }
}

// ---------- causal flash attention, summed over M sources ----------
__global__ __launch_bounds__(128) void attn_kernel(const bfraw* __restrict__ qb,
                                                   const bfraw* __restrict__ kbuf,
                                                   const bfraw* __restrict__ vbuf,
                                                   bfraw* __restrict__ ob) {
  __shared__ __align__(16) bfraw ldsP[4][16 * 32];
  const int lane = threadIdx.x & 31;
  const int wid  = threadIdx.x >> 5;
  const int h = blockIdx.y, b = blockIdx.z;
  const int q0 = (blockIdx.x * 4 + wid) * 16;
  const size_t bhOff    = (size_t)(b * Hh + h) * Ss * HSz;
  const size_t kvStride = (size_t)Bb * Hh * Ss * HSz;
  const bfraw* Q = qb + bhOff;
  bfraw* lds = ldsP[wid];

  const int n  = lane & 15;
  const int rb = (lane < 16) ? 0 : 8;

  v16bf qa0 = load_a(Q, HSz, q0, 0, lane);
  v16bf qa1 = load_a(Q, HSz, q0, 32, lane);

  float accT[4][8];
#pragma unroll
  for (int t = 0; t < 4; ++t)
#pragma unroll
    for (int r = 0; r < 8; ++r) accT[t][r] = 0.f;

  for (int m = 0; m < Mm; ++m) {
    const bfraw* K = kbuf + (size_t)m * kvStride + bhOff;
    const bfraw* V = vbuf + (size_t)m * kvStride + bhOff;
    float mrow[8], lrow[8];
    v8f o[4] = {};
#pragma unroll
    for (int r = 0; r < 8; ++r) { mrow[r] = -1e30f; lrow[r] = 0.f; }

    for (int kt = 0; kt <= q0 + 15; kt += 32) {  // causal chunk skipping
      // prefetch next chunk's K rows / V panel (global_prefetch_b8) while
      // this chunk's WMMAs + softmax execute
      if (kt + 32 <= q0 + 15) {
        __builtin_prefetch(K + (size_t)(kt + 32 + lane) * HSz, 0, 3);
        __builtin_prefetch(V + (size_t)((kt + 32) >> 5) * (HSz * 32) + lane * 64,
                           0, 3);
      }
      v8f s0 = {}, s1 = {};
      s0 = wmma_bf16(qa0, load_kT(K, kt, 0, lane), s0);
      s0 = wmma_bf16(qa1, load_kT(K, kt, 32, lane), s0);
      s1 = wmma_bf16(qa0, load_kT(K, kt + 16, 0, lane), s1);
      s1 = wmma_bf16(qa1, load_kT(K, kt + 16, 32, lane), s1);

#pragma unroll
      for (int r = 0; r < 8; ++r) {
        const int qrow = q0 + rb + r;
        float sc0 = (kt + n      <= qrow) ? s0[r] * 0.125f : -1e30f;
        float sc1 = (kt + 16 + n <= qrow) ? s1[r] * 0.125f : -1e30f;
        float t = fmaxf(sc0, sc1);
        t = fmaxf(t, __shfl_xor(t, 1, 16));
        t = fmaxf(t, __shfl_xor(t, 2, 16));
        t = fmaxf(t, __shfl_xor(t, 4, 16));
        t = fmaxf(t, __shfl_xor(t, 8, 16));
        const float nm    = fmaxf(mrow[r], t);
        const float alpha = __expf(mrow[r] - nm);
        mrow[r] = nm;
        float p0 = __expf(sc0 - nm);
        float p1 = __expf(sc1 - nm);
        float ps = p0 + p1;
        ps += __shfl_xor(ps, 1, 16);
        ps += __shfl_xor(ps, 2, 16);
        ps += __shfl_xor(ps, 4, 16);
        ps += __shfl_xor(ps, 8, 16);
        lrow[r] = lrow[r] * alpha + ps;
#pragma unroll
        for (int t4 = 0; t4 < 4; ++t4) o[t4][r] *= alpha;
        lds[(rb + r) * 32 + n]      = f2br(p0);
        lds[(rb + r) * 32 + 16 + n] = f2br(p1);
      }
      // reload probs as A fragment (two ds_load_b128; same-wave DS in-order)
      v16bf pa = frag_c(&lds[(lane & 15) * 32 + ((lane < 16) ? 0 : 16)]);
#pragma unroll
      for (int t4 = 0; t4 < 4; ++t4)
        o[t4] = wmma_bf16(pa, load_vp(V, kt, t4 * 16, lane), o[t4]);
    }
#pragma unroll
    for (int r = 0; r < 8; ++r) {
      const float inv = 1.f / lrow[r];
#pragma unroll
      for (int t4 = 0; t4 < 4; ++t4) accT[t4][r] += o[t4][r] * inv;
    }
  }
#pragma unroll
  for (int t4 = 0; t4 < 4; ++t4)
#pragma unroll
    for (int r = 0; r < 8; ++r)
      ob[((size_t)b * Ss + q0 + rb + r) * DM + h * HSz + t4 * 16 + n] =
          f2br(accT[t4][r]);
}

// ---------- bf16-A x packed-bf16-W GEMM + bias (+tanh); out f32 or bf16 ----------
__global__ __launch_bounds__(256) void gemm_kernel(const bfraw* __restrict__ A,
                                                   const bfraw* __restrict__ Wp,
                                                   const float* __restrict__ bias,
                                                   float* __restrict__ outF,
                                                   bfraw* __restrict__ outB,
                                                   int Mrows, int Kdim, int N,
                                                   int doTanh) {
  const int lane = threadIdx.x & 31;
  const int wid  = threadIdx.x >> 5;
  const int tile = blockIdx.x * (blockDim.x >> 5) + wid;
  const int ct = tile % (N / 64);
  const int rt = tile / (N / 64);
  if (rt * 16 >= Mrows) return;  // whole-wave exit; EXEC all-ones for WMMA
  const int row0 = rt * 16, col0 = ct * 64;

  v8f acc[4] = {};
  for (int k0 = 0; k0 < Kdim; k0 += 32) {
    v16bf a = load_a(A, Kdim, row0, k0, lane);
#pragma unroll
    for (int t = 0; t < 4; ++t)
      acc[t] = wmma_bf16(a, load_bp(Wp, N, k0, col0 + t * 16, lane), acc[t]);
  }
  const int n  = lane & 15;
  const int rb = (lane < 16) ? 0 : 8;
#pragma unroll
  for (int t = 0; t < 4; ++t)
#pragma unroll
    for (int r = 0; r < 8; ++r) {
      const int col = col0 + t * 16 + n;
      float v = acc[t][r] + bias[col];
      if (doTanh) v = tanhf(v);
      const size_t idx = (size_t)(row0 + rb + r) * N + col;
      if (outF) outF[idx] = v;
      else      outB[idx] = f2br(v);
    }
}

extern "C" void kernel_launch(void* const* d_in, const int* in_sizes, int n_in,
                              void* d_out, int out_size, void* d_ws, size_t ws_size,
                              hipStream_t stream) {
  const float* query_x = (const float*)d_in[0];
  const float* kv_x    = (const float*)d_in[1];
  const float* Wq      = (const float*)d_in[2];
  const float* Wkv     = (const float*)d_in[3];
  const float* W1      = (const float*)d_in[4];
  const float* b1      = (const float*)d_in[5];
  const float* W2      = (const float*)d_in[6];
  const float* b2      = (const float*)d_in[7];
  (void)in_sizes; (void)n_in; (void)out_size; (void)ws_size;

  const size_t xqN  = (size_t)Bb * Ss * Cc;            // 8.4M
  const size_t xkvN = (size_t)Mm * xqN;                // 25.2M
  const size_t qN   = (size_t)Bb * Hh * Ss * HSz;      // 8.4M

  bfraw* xq   = (bfraw*)d_ws;
  bfraw* xkv  = xq   + xqN;
  bfraw* qbuf = xkv  + xkvN;
  bfraw* kbuf = qbuf + qN;
  bfraw* vbuf = kbuf + (size_t)Mm * qN;
  bfraw* obuf = vbuf + (size_t)Mm * qN;                // (B*S, DM)
  bfraw* hbuf = obuf + (size_t)Bb * Ss * DM;           // (B*S, CH)
  bfraw* wqp  = hbuf + (size_t)Bb * Ss * CH;           // H * C * HS
  bfraw* wkp  = wqp  + (size_t)Hh * Cc * HSz;          // M*H * C * HS
  bfraw* wvp  = wkp  + (size_t)Mm * Hh * Cc * HSz;
  bfraw* w1p  = wvp  + (size_t)Mm * Hh * Cc * HSz;     // DM * CH
  bfraw* w2p  = w1p  + (size_t)DM * CH;                // CH * DM

  // 1) pack activations to bf16
  pack_x_kernel<<<(int)(xqN / 8 / 256), 256, 0, stream>>>(query_x, xq, (long)xqN);
  pack_x_kernel<<<(int)(xkvN / 8 / 256), 256, 0, stream>>>(kv_x, xkv, (long)xkvN);

  // 2) pack weights into K-panel bf16 layout
  pack_w_kernel<<<dim3(Cc * HSz / 256, Hh), 256, 0, stream>>>(
      Wq, wqp, HSz, 0, Cc, HSz, (long)Cc * HSz, (long)Cc * HSz);
  pack_w_kernel<<<dim3(Cc * HSz / 256, Mm * Hh), 256, 0, stream>>>(
      Wkv, wkp, 2 * HSz, 0, Cc, HSz, (long)Cc * 2 * HSz, (long)Cc * HSz);
  pack_w_kernel<<<dim3(Cc * HSz / 256, Mm * Hh), 256, 0, stream>>>(
      Wkv, wvp, 2 * HSz, HSz, Cc, HSz, (long)Cc * 2 * HSz, (long)Cc * HSz);
  pack_w_kernel<<<dim3(DM * CH / 256, 1), 256, 0, stream>>>(
      W1, w1p, CH, 0, DM, CH, 0, 0);
  pack_w_kernel<<<dim3(CH * DM / 256, 1), 256, 0, stream>>>(
      W2, w2p, DM, 0, CH, DM, 0, 0);

  // 3) projections (V written in key-panel layout for the P*V WMMAs)
  dim3 pg(Ss / 64, Hh, Bb);
  proj_kernel<<<pg, 128, 0, stream>>>(xq, wqp, qbuf, 0);
  for (int m = 0; m < Mm; ++m) {
    const bfraw* Xm = xkv + (size_t)m * xqN;
    proj_kernel<<<pg, 128, 0, stream>>>(Xm, wkp + (size_t)m * Hh * Cc * HSz,
                                        kbuf + (size_t)m * qN, 0);
    proj_kernel<<<pg, 128, 0, stream>>>(Xm, wvp + (size_t)m * Hh * Cc * HSz,
                                        vbuf + (size_t)m * qN, 1);
  }

  // 4) attention
  attn_kernel<<<pg, 128, 0, stream>>>(qbuf, kbuf, vbuf, obuf);

  // 5) MLP
  const int rows = Bb * Ss;  // 8192
  gemm_kernel<<<(rows / 16) * (CH / 64) / 8, 256, 0, stream>>>(
      obuf, w1p, b1, nullptr, hbuf, rows, DM, CH, 1);
  gemm_kernel<<<(rows / 16) * (Cc / 64) / 8, 256, 0, stream>>>(
      hbuf, w2p, b2, (float*)d_out, nullptr, rows, CH, Cc, 0);
}
